// SGNS_46832323396194
// MI455X (gfx1250) — matrix operand order repaired and checked by reading, not compiled, not run on hardware
//
#include <hip/hip_runtime.h>
#include <hip/hip_bf16.h>

typedef __attribute__((ext_vector_type(16))) _Float16 v16h;
typedef __attribute__((ext_vector_type(8)))  float    v8f;

#define EMBED 128
#define KNEG  10
#define BSZ   65536
#define EPSV  1e-10f
#define WAVES_PER_BLOCK 8   // 256 threads = 8 wave32

// ---- A-matrix (16x32 f16, M x K) per-lane loader -------------------------
// ISA layout (CDNA5 ISA): lane L, h = L>>4, row M = L&15.
//   v16h element e in [0,7]  -> K = kbase + h*8 + e
//   v16h element e in [8,15] -> K = kbase + 16 + h*8 + (e-8)
// Each run of 8 floats is 32B-aligned (kbase multiple of 32, h*8 floats).
__device__ __forceinline__ v16h load_A_chunk(const float* __restrict__ row,
                                             int kbase, int h) {
  const float4* p0 = reinterpret_cast<const float4*>(row + kbase + h * 8);
  const float4* p1 = reinterpret_cast<const float4*>(row + kbase + 16 + h * 8);
  float4 f0 = p0[0], f1 = p0[1], f2 = p1[0], f3 = p1[1];
  v16h a;
  a[0]  = (_Float16)f0.x; a[1]  = (_Float16)f0.y;
  a[2]  = (_Float16)f0.z; a[3]  = (_Float16)f0.w;
  a[4]  = (_Float16)f1.x; a[5]  = (_Float16)f1.y;
  a[6]  = (_Float16)f1.z; a[7]  = (_Float16)f1.w;
  a[8]  = (_Float16)f2.x; a[9]  = (_Float16)f2.y;
  a[10] = (_Float16)f2.z; a[11] = (_Float16)f2.w;
  a[12] = (_Float16)f3.x; a[13] = (_Float16)f3.y;
  a[14] = (_Float16)f3.z; a[15] = (_Float16)f3.w;
  return a;
}

// ---- B-matrix (32x16 f16, K x N) per-lane loader -------------------------
// Lane L: column N = L&15, group g = L>>4.
//   v16h element e in [0,7]  -> K = kbase + g*8 + e
//   v16h element e in [8,15] -> K = kbase + 16 + g*8 + (e-8)
__device__ __forceinline__ v16h load_B_chunk(const float* __restrict__ row,
                                             int kbase, int g) {
  const float4* p0 = reinterpret_cast<const float4*>(row + kbase + g * 8);
  const float4* p1 = reinterpret_cast<const float4*>(row + kbase + 16 + g * 8);
  float4 f0 = p0[0], f1 = p0[1], f2 = p1[0], f3 = p1[1];
  v16h b;
  b[0]  = (_Float16)f0.x; b[1]  = (_Float16)f0.y;
  b[2]  = (_Float16)f0.z; b[3]  = (_Float16)f0.w;
  b[4]  = (_Float16)f1.x; b[5]  = (_Float16)f1.y;
  b[6]  = (_Float16)f1.z; b[7]  = (_Float16)f1.w;
  b[8]  = (_Float16)f2.x; b[9]  = (_Float16)f2.y;
  b[10] = (_Float16)f2.z; b[11] = (_Float16)f2.w;
  b[12] = (_Float16)f3.x; b[13] = (_Float16)f3.y;
  b[14] = (_Float16)f3.z; b[15] = (_Float16)f3.w;
  return b;
}

__global__ void sgns_zero_kernel(float* __restrict__ out) {
  if (threadIdx.x == 0 && blockIdx.x == 0) out[0] = 0.0f;
}

// One wave32 per tile of 16 batch items. 11 diag-of-GEMM chains per tile,
// each chain = 4x v_wmma_f32_16x16x32_f16 over K=128.
__global__ __launch_bounds__(256)
void sgns_wmma_kernel(const float* __restrict__ tw,   // [VOCAB, 128]
                      const float* __restrict__ cw,   // [VOCAB, 128]
                      const int*   __restrict__ t,    // [B]
                      const int*   __restrict__ c,    // [B]
                      const int*   __restrict__ n,    // [B, 10]
                      float* __restrict__ out) {
  const int lane = threadIdx.x & 31;
  const int wave = threadIdx.x >> 5;
  const int tile = blockIdx.x * WAVES_PER_BLOCK + wave; // 4096 tiles
  const int m    = lane & 15;        // batch item within tile / matrix row
  const int h    = lane >> 4;        // lane group within wave
  const int b    = tile * 16 + m;    // this lane's batch element

  // ---- A operand: 16 target rows, loaded once, reused across 11 contexts
  const float* vt = tw + (size_t)t[b] * EMBED;
  v16h A0 = load_A_chunk(vt,  0, h);
  v16h A1 = load_A_chunk(vt, 32, h);
  v16h A2 = load_A_chunk(vt, 64, h);
  v16h A3 = load_A_chunk(vt, 96, h);

  // Diagonal ownership (C/D layout): lanes 0..7 -> acc[lane] is D[lane][lane];
  // lanes 24..31 -> acc[lane-24] is D[lane-16][lane-16]. Others: none.
  const int sel = (lane < 8) ? lane : ((lane >= 24) ? (lane - 24) : -1);

  float local = 0.0f;

  for (int k = 0; k <= KNEG; ++k) {
    // Wave-uniform selection of this round's context row (EXEC stays full).
    int row;
    if (k == 0) row = c[b];
    else        row = n[(size_t)b * KNEG + (k - 1)];
    const float* vc = cw + (size_t)row * EMBED;

    // Prefetch next negative's row while this WMMA chain runs
    // (lowers to global_prefetch_b8). Two lines per lane, 2 lanes/row.
    if (k < KNEG) {
      int nrow = n[(size_t)b * KNEG + k];
      const float* np = cw + (size_t)nrow * EMBED + h * 64;
      __builtin_prefetch(np, 0, 1);
      __builtin_prefetch(np + 32, 0, 1);
    }

    v16h B0 = load_B_chunk(vc,  0, h);
    v16h B1 = load_B_chunk(vc, 32, h);
    v16h B2 = load_B_chunk(vc, 64, h);
    v16h B3 = load_B_chunk(vc, 96, h);

    v8f acc = {};
    acc = __builtin_amdgcn_wmma_f32_16x16x32_f16(false, A0, false, B0,
                                                 (short)0, acc, false, false);
    acc = __builtin_amdgcn_wmma_f32_16x16x32_f16(false, A1, false, B1,
                                                 (short)0, acc, false, false);
    acc = __builtin_amdgcn_wmma_f32_16x16x32_f16(false, A2, false, B2,
                                                 (short)0, acc, false, false);
    acc = __builtin_amdgcn_wmma_f32_16x16x32_f16(false, A3, false, B3,
                                                 (short)0, acc, false, false);

    // Extract this lane's diagonal element (branch-free).
    float score = 0.0f;
#pragma unroll
    for (int i = 0; i < 8; ++i) score = (sel == i) ? acc[i] : score;

    // k==0: -log(sigmoid(+s)+eps); k>0: -log(sigmoid(-s)+eps)
    float x    = (k == 0) ? score : -score;
    float sig  = 1.0f / (1.0f + __expf(-x));
    float term = -__logf(sig + EPSV);
    local += (sel >= 0) ? term : 0.0f;   // inactive lanes contribute 0
  }

  // Wave32 reduction of the 16 active diag lanes' partial sums.
#pragma unroll
  for (int off = 16; off > 0; off >>= 1)
    local += __shfl_xor(local, off, 32);

  if (lane == 0)
    atomicAdd(out, local * (1.0f / (float)BSZ));
}

extern "C" void kernel_launch(void* const* d_in, const int* in_sizes, int n_in,
                              void* d_out, int out_size, void* d_ws, size_t ws_size,
                              hipStream_t stream) {
  const float* tw = (const float*)d_in[0];  // t_weight [100000,128] f32
  const float* cw = (const float*)d_in[1];  // c_weight [100000,128] f32
  const int*   t  = (const int*)d_in[2];    // [65536]
  const int*   c  = (const int*)d_in[3];    // [65536]
  const int*   n  = (const int*)d_in[4];    // [65536,10]
  float* out = (float*)d_out;

  sgns_zero_kernel<<<1, 1, 0, stream>>>(out);

  const int tiles  = BSZ / 16;                 // 4096 waves
  const int blocks = tiles / WAVES_PER_BLOCK;  // 512 blocks of 256 threads
  sgns_wmma_kernel<<<blocks, 256, 0, stream>>>(tw, cw, t, c, n, out);
}
